// SPINN_58884001628711
// MI455X (gfx1250) — matrix operand order repaired and checked by reading, not compiled, not run on hardware
//
#include <hip/hip_runtime.h>
#include <hip/hip_bf16.h>

// ---------------- problem constants (from reference) ----------------
#define Lc    128
#define Bc    512
#define Dc    512
#define Mc    256          // hidden size = D/2
#define NG    1280         // 5*Mc gate columns
#define DEPTH (Lc + 2)     // stack depth

// ---------------- CDNA5 WMMA types ----------------
typedef __attribute__((ext_vector_type(16))) __bf16 v16bf;
typedef __attribute__((ext_vector_type(8)))  __bf16 v8bf;
typedef __attribute__((ext_vector_type(8)))  float  v8f;

__device__ __forceinline__ unsigned short f2bf(float f) {
    unsigned u = __float_as_uint(f);
    u += 0x7fffu + ((u >> 16) & 1u);     // round-to-nearest-even
    return (unsigned short)(u >> 16);
}
__device__ __forceinline__ float sigm(float x) { return 1.0f / (1.0f + __expf(-x)); }

__device__ __forceinline__ v16bf load_afrag(const unsigned short* p) {
    v8bf lo = *(const v8bf*)(p);
    v8bf hi = *(const v8bf*)(p + 16);
    return __builtin_shufflevector(lo, hi, 0,1,2,3,4,5,6,7,8,9,10,11,12,13,14,15);
}

// ---------------- init: stack[:,0]=stack[:,1]=buffers[0], sp=2, bp=L ----------------
__global__ void spinn_init(const float* __restrict__ buffers, float* __restrict__ stack,
                           int* __restrict__ sp, int* __restrict__ bp) {
    int tid = blockIdx.x * blockDim.x + threadIdx.x;
    if (tid >= Bc * Dc) return;
    int b = tid / Dc, d = tid % Dc;
    float v = buffers[(size_t)b * Dc + d];                 // l = 0 row
    stack[((size_t)b * DEPTH + 0) * Dc + d] = v;
    stack[((size_t)b * DEPTH + 1) * Dc + d] = v;
    if (d == 0) { sp[b] = 2; bp[b] = Lc; }
}

// ---------------- repack weights into WMMA bf16 B-fragment layout ----------------
// Bp element ((kt*NG + n)*2 + h)*16 + j  holds  W[(kt*32 + 16*(j>>3) + 8*h + (j&7))*NG + n]
// -> each lane reads one contiguous, 32B-aligned v16bf per fragment.
__global__ void spinn_pack_w(const float* __restrict__ Wl, const float* __restrict__ Wr,
                             unsigned short* __restrict__ Bpl, unsigned short* __restrict__ Bpr) {
    int tid = blockIdx.x * blockDim.x + threadIdx.x;
    const int per = 8 * NG * 32;                 // 327680 per matrix
    if (tid >= 2 * per) return;
    int mat = tid / per, rem = tid % per;
    int kt = rem / (NG * 32);
    int r2 = rem % (NG * 32);
    int n  = r2 / 32;
    int q  = r2 % 32;                            // q = h*16 + j
    int k  = kt * 32 + (q & 7) + ((q >> 4) << 3) + (((q >> 3) & 1) << 4);
    const float* src = mat ? Wr : Wl;
    unsigned short* dst = mat ? Bpr : Bpl;
    dst[rem] = f2bf(src[(size_t)k * NG + n]);
}

// ---------------- per-step: gather stack tops -> bf16 A matrices ----------------
__global__ void spinn_prep(const float* __restrict__ stack, const int* __restrict__ sp,
                           unsigned short* __restrict__ Al, unsigned short* __restrict__ Ar) {
    int tid = blockIdx.x * blockDim.x + threadIdx.x;
    if (tid >= Bc * Mc) return;
    int b = tid / Mc, m = tid % Mc;
    int s  = sp[b];
    int pl = s - 2 > 0 ? s - 2 : 0;
    int pr = s - 1 > 0 ? s - 1 : 0;
    Al[tid] = f2bf(stack[((size_t)b * DEPTH + pl) * Dc + m]);
    Ar[tid] = f2bf(stack[((size_t)b * DEPTH + pr) * Dc + m]);
}

// ---------------- fused dual-GEMM + TreeLSTM epilogue (WMMA bf16) ----------------
// 1 wave per block; each wave computes a 32-row (2 row tiles) x 16-hidden-col slab
// across all 5 gate blocks: acc[2][5], 160 v_wmma_f32_16x16x32_bf16 per wave.
// All A/B fragment loads are immediate offsets from per-wave base pointers
// (max offset ~640KB, fits the signed 24-bit IOFFSET field).
__global__ void __launch_bounds__(32)
spinn_reduce(const unsigned short* __restrict__ Al, const unsigned short* __restrict__ Ar,
             const unsigned short* __restrict__ Bpl, const unsigned short* __restrict__ Bpr,
             const float* __restrict__ b_left, float* __restrict__ stack,
             const int* __restrict__ sp, const int* __restrict__ trans_t) {
    const int lane = threadIdx.x;
    const int cl   = lane & 15;     // A row within tile / B+D column within tile
    const int h    = lane >> 4;     // half-wave select
    const int r0   = blockIdx.x * 32;
    const int j    = blockIdx.y;

    v8f acc[2][5];
#pragma unroll
    for (int rt = 0; rt < 2; ++rt)
#pragma unroll
        for (int g = 0; g < 5; ++g) acc[rt][g] = (v8f)(0.0f);

    // per-wave base pointers; everything below is base + compile-time offset
    const unsigned short* aL0 = Al + (size_t)(r0 + cl) * Mc + h * 8;
    const unsigned short* aR0 = Ar + (size_t)(r0 + cl) * Mc + h * 8;
    const unsigned short* bL0 = Bpl + ((size_t)(j * 16 + cl) * 2 + h) * 16;
    const unsigned short* bR0 = Bpr + ((size_t)(j * 16 + cl) * 2 + h) * 16;

#pragma unroll
    for (int kt = 0; kt < 8; ++kt) {
        const int ao = kt * 32;                      // A: 64B per K-chunk
        const int bo = kt * (NG * 32);               // B: 80KB per K-chunk
        v16bf aL0f = load_afrag(aL0 + ao);
        v16bf aL1f = load_afrag(aL0 + 16 * Mc + ao);
        v16bf aR0f = load_afrag(aR0 + ao);
        v16bf aR1f = load_afrag(aR0 + 16 * Mc + ao);
#pragma unroll
        for (int g = 0; g < 5; ++g) {
            v16bf bLf = *(const v16bf*)(bL0 + bo + g * (Mc * 32));   // +16KB per gate
            v16bf bRf = *(const v16bf*)(bR0 + bo + g * (Mc * 32));
            acc[0][g] = __builtin_amdgcn_wmma_f32_16x16x32_bf16(
                            false, aL0f, false, bLf, (short)0, acc[0][g], false, false);
            acc[0][g] = __builtin_amdgcn_wmma_f32_16x16x32_bf16(
                            false, aR0f, false, bRf, (short)0, acc[0][g], false, false);
            acc[1][g] = __builtin_amdgcn_wmma_f32_16x16x32_bf16(
                            false, aL1f, false, bLf, (short)0, acc[1][g], false, false);
            acc[1][g] = __builtin_amdgcn_wmma_f32_16x16x32_bf16(
                            false, aR1f, false, bRf, (short)0, acc[1][g], false, false);
        }
    }

    // epilogue: C/D layout -> lane holds col cl, rows v + 8*h for VGPR v
    const int m = j * 16 + cl;      // hidden unit index 0..255
#pragma unroll
    for (int rt = 0; rt < 2; ++rt) {
#pragma unroll
        for (int v = 0; v < 8; ++v) {
            int b = r0 + rt * 16 + v + 8 * h;
            if (trans_t[b] == 2) {
                int s  = sp[b];
                int pl = s - 2 > 0 ? s - 2 : 0;
                int pr = s - 1 > 0 ? s - 1 : 0;
                float a  = acc[rt][0][v] + b_left[m];
                float ig = acc[rt][1][v] + b_left[Mc + m];
                float f1 = acc[rt][2][v] + b_left[2 * Mc + m];
                float f2 = acc[rt][3][v] + b_left[3 * Mc + m];
                float og = acc[rt][4][v] + b_left[4 * Mc + m];
                float c_l = stack[((size_t)b * DEPTH + pl) * Dc + Mc + m];
                float c_r = stack[((size_t)b * DEPTH + pr) * Dc + Mc + m];
                float cc  = tanhf(a) * sigm(ig) + sigm(f1) * c_l + sigm(f2) * c_r;
                float hh  = sigm(og) * tanhf(cc);
                stack[((size_t)b * DEPTH + pl) * Dc + m]      = hh;
                stack[((size_t)b * DEPTH + pl) * Dc + Mc + m] = cc;
            }
        }
    }
}

// ---------------- per-step: masked shift copy (buffers row -> stack[sp]) ----------------
__global__ void spinn_shift(const float* __restrict__ buffers, const int* __restrict__ trans_t,
                            const int* __restrict__ sp, const int* __restrict__ bp,
                            float* __restrict__ stack) {
    int tid = blockIdx.x * blockDim.x + threadIdx.x;
    if (tid >= Bc * Dc) return;
    int b = tid / Dc, d = tid % Dc;
    if (trans_t[b] == 1) {
        int p  = sp[b];
        int bl = bp[b] - 1 > 0 ? bp[b] - 1 : 0;
        stack[((size_t)b * DEPTH + p) * Dc + d] = buffers[((size_t)bl * Bc + b) * Dc + d];
    }
}

// ---------------- per-step: pointer update ----------------
__global__ void spinn_update(const int* __restrict__ trans_t, int* __restrict__ sp,
                             int* __restrict__ bp) {
    int b = blockIdx.x * blockDim.x + threadIdx.x;
    if (b >= Bc) return;
    int tr = trans_t[b];
    sp[b] += (tr == 1) - (tr == 2);
    bp[b] -= (tr == 1);
}

// ---------------- output: stack[b, sp-1, :Mc] ----------------
__global__ void spinn_out(const float* __restrict__ stack, const int* __restrict__ sp,
                          float* __restrict__ out) {
    int tid = blockIdx.x * blockDim.x + threadIdx.x;
    if (tid >= Bc * Mc) return;
    int b = tid / Mc, m = tid % Mc;
    int p = sp[b] - 1 > 0 ? sp[b] - 1 : 0;
    out[tid] = stack[((size_t)b * DEPTH + p) * Dc + m];
}

extern "C" void kernel_launch(void* const* d_in, const int* in_sizes, int n_in,
                              void* d_out, int out_size, void* d_ws, size_t ws_size,
                              hipStream_t stream) {
    const float* buffers     = (const float*)d_in[0];
    const int*   transitions = (const int*)d_in[1];   // (T, B) int32 on device
    const float* W_left      = (const float*)d_in[2];
    const float* b_left      = (const float*)d_in[3];
    const float* W_right     = (const float*)d_in[4];
    float* out = (float*)d_out;

    const int T = in_sizes[1] / Bc;                   // 253

    // workspace layout (all offsets 32B-aligned)
    char* ws = (char*)d_ws;
    const size_t stackBytes = (size_t)Bc * DEPTH * Dc * sizeof(float);      // 136,314,880
    const size_t bpBytes    = (size_t)8 * NG * 32 * sizeof(unsigned short); // 655,360
    const size_t aBytes     = (size_t)Bc * Mc * sizeof(unsigned short);     // 262,144
    float*          stack = (float*)ws;
    unsigned short* Bpl   = (unsigned short*)(ws + stackBytes);
    unsigned short* Bpr   = (unsigned short*)(ws + stackBytes + bpBytes);
    unsigned short* Al    = (unsigned short*)(ws + stackBytes + 2 * bpBytes);
    unsigned short* Ar    = (unsigned short*)(ws + stackBytes + 2 * bpBytes + aBytes);
    int*            sp    = (int*)(ws + stackBytes + 2 * bpBytes + 2 * aBytes);
    int*            bp    = (int*)(ws + stackBytes + 2 * bpBytes + 2 * aBytes + Bc * sizeof(int));

    const dim3 blk256(256);
    spinn_init<<<dim3((Bc * Dc + 255) / 256), blk256, 0, stream>>>(buffers, stack, sp, bp);
    spinn_pack_w<<<dim3((2 * 8 * NG * 32 + 255) / 256), blk256, 0, stream>>>(W_left, W_right, Bpl, Bpr);

    for (int t = 0; t < T; ++t) {
        const int* trans_t = transitions + (size_t)t * Bc;
        spinn_prep<<<dim3((Bc * Mc + 255) / 256), blk256, 0, stream>>>(stack, sp, Al, Ar);
        spinn_reduce<<<dim3(Bc / 32, Mc / 16), dim3(32), 0, stream>>>(
            Al, Ar, Bpl, Bpr, b_left, stack, sp, trans_t);
        spinn_shift<<<dim3((Bc * Dc + 255) / 256), blk256, 0, stream>>>(buffers, trans_t, sp, bp, stack);
        spinn_update<<<dim3((Bc + 255) / 256), blk256, 0, stream>>>(trans_t, sp, bp);
    }

    spinn_out<<<dim3((Bc * Mc + 255) / 256), blk256, 0, stream>>>(stack, sp, out);
}
